// TransformerRefine_2808908611848
// MI455X (gfx1250) — matrix operand order repaired and checked by reading, not compiled, not run on hardware
//
#include <hip/hip_runtime.h>
#include <hip/hip_bf16.h>
#include <math.h>

typedef __attribute__((ext_vector_type(16))) _Float16 v16h;
typedef __attribute__((ext_vector_type(8)))  _Float16 v8h;
typedef __attribute__((ext_vector_type(8)))  float    v8f;

#define NBATCH 8
#define CIN    256
#define NTOK   1024
#define DMODEL 512
#define NDEPTH 6
#define NHEAD  8
#define HDIM   64
#define HIDDEN 2048
#define NCOUT  256
#define MTOT   8192   // B * N tokens

// ---------------------------------------------------------------------------
// WMMA helpers (CDNA5 wave32, v_wmma_f32_16x16x32_f16)
// ---------------------------------------------------------------------------
__device__ __forceinline__ v8f wmma32(v16h a, v16h b, v8f c) {
  return __builtin_amdgcn_wmma_f32_16x16x32_f16(false, a, false, b, (short)0, c,
                                                false, false);
}

// A fragment (16x32 f16): lane m = l&15; k = ((l>>4)<<4) + i
__device__ __forceinline__ v16h load_afrag(const _Float16* base, int lda,
                                           int m0, int k0) {
  int lane = threadIdx.x & 31;
  const _Float16* p =
      base + (size_t)(m0 + (lane & 15)) * lda + k0 + ((lane >> 4) << 4);
  v8h lo = *(const v8h*)(p);
  v8h hi = *(const v8h*)(p + 8);
  v16h a;
#pragma unroll
  for (int i = 0; i < 8; ++i) { a[i] = lo[i]; a[i + 8] = hi[i]; }
  return a;
}

// B fragment (32x16 f16) from row-major [N,K]: B[k][n] = W[n][k]
// lane: n = l&15 (row of W), k = ((l>>4)<<4)+i (16 contiguous halves)
__device__ __forceinline__ v16h load_bfrag(const _Float16* base, int ldb,
                                           int n0, int k0) {
  int lane = threadIdx.x & 31;
  const _Float16* p =
      base + (size_t)(n0 + (lane & 15)) * ldb + k0 + ((lane >> 4) << 4);
  return *(const v16h*)p;
}

// ---------------------------------------------------------------------------
// WMMA GEMM: C[M,N] = A[M,K](f16) * W[N,K]^T(f16), fused epilogues.
// 8 waves / block; each wave computes a 64x64 tile = 4x4 WMMA tiles.
// When LDSA: all 8 waves of a block share one 64-row A tile; it is staged
// into double-buffered LDS with GLOBAL_LOAD_ASYNC_TO_LDS_B128 (ASYNCcnt),
// overlapping the copy of step s+1 with the 16 WMMAs of step s.
// ---------------------------------------------------------------------------
// EPI 0: store f16                          (qkv)
// EPI 1: +bias, exact GELU, store f16       (fc1)
// EPI 2: +bias +extra(resid f32), store f32 (proj / fc2 -> residual stream)
// EPI 3: +bias +pos[row&1023], store f32    (input projection)
// EPI 4: +bias, store f32 as BCHW           (output projection)
template <int EPI, bool LDSA>
__global__ __launch_bounds__(256) void gemm_kernel(
    const _Float16* __restrict__ A, const _Float16* __restrict__ W,
    const float* __restrict__ bias, const float* __restrict__ extra,
    _Float16* __restrict__ outh, float* __restrict__ outf,
    int M, int N, int K) {
  __shared__ __align__(16) _Float16 aTile[2][64 * 32];  // 2 x 4KB
  int lane = threadIdx.x & 31;
  int wave = blockIdx.x * 8 + (threadIdx.x >> 5);
  int ntiles = N >> 6;
  int mt = wave / ntiles;
  int nt = wave - mt * ntiles;
  int m0 = mt << 6, n0 = nt << 6;

  // cooperative async staging: 256 threads x 16B = 64x32 f16 tile
  int srow = threadIdx.x >> 2, schunk = (threadIdx.x & 3) << 3;
  auto issue_stage = [&](int buf, int k0) {
    const _Float16* g = A + (size_t)(m0 + srow) * K + k0 + schunk;
    unsigned l = (unsigned)(uintptr_t)&aTile[buf][srow * 32 + schunk];
    asm volatile("global_load_async_to_lds_b128 %0, %1, off"
                 :: "v"(l), "v"(g) : "memory");
  };

  v8f zero = {};
  v8f acc[4][4];
#pragma unroll
  for (int mi = 0; mi < 4; ++mi)
#pragma unroll
    for (int ni = 0; ni < 4; ++ni) acc[mi][ni] = zero;

  int nsteps = K >> 5;
  if (LDSA) issue_stage(0, 0);

  for (int s = 0; s < nsteps; ++s) {
    int k0 = s << 5;
    v16h af[4], bf[4];
    if (LDSA) {
      if (s + 1 < nsteps) {
        issue_stage((s + 1) & 1, k0 + 32);
        asm volatile("s_wait_asynccnt 0x1" ::: "memory");
      } else {
        asm volatile("s_wait_asynccnt 0x0" ::: "memory");
      }
      __syncthreads();
      const _Float16* at = aTile[s & 1];
#pragma unroll
      for (int mi = 0; mi < 4; ++mi) af[mi] = load_afrag(at, 32, mi * 16, 0);
    } else {
      __builtin_prefetch(A + (size_t)(m0 + (lane & 15)) * K + k0 + 64, 0, 0);
#pragma unroll
      for (int mi = 0; mi < 4; ++mi)
        af[mi] = load_afrag(A, K, m0 + mi * 16, k0);
    }
    __builtin_prefetch(W + (size_t)(n0 + (lane & 15)) * K + k0 + 64, 0, 0);
#pragma unroll
    for (int ni = 0; ni < 4; ++ni) bf[ni] = load_bfrag(W, K, n0 + ni * 16, k0);
#pragma unroll
    for (int mi = 0; mi < 4; ++mi)
#pragma unroll
      for (int ni = 0; ni < 4; ++ni)
        acc[mi][ni] = wmma32(af[mi], bf[ni], acc[mi][ni]);
    if (LDSA) __syncthreads();  // reads of this buffer done before re-stage
  }

#pragma unroll
  for (int mi = 0; mi < 4; ++mi) {
#pragma unroll
    for (int ni = 0; ni < 4; ++ni) {
#pragma unroll
      for (int r = 0; r < 8; ++r) {
        int row = m0 + mi * 16 + ((lane >> 4) << 3) + r;
        int col = n0 + ni * 16 + (lane & 15);
        float v = acc[mi][ni][r];
        if (EPI == 0) {
          outh[(size_t)row * N + col] = (_Float16)v;
        } else if (EPI == 1) {
          v += bias[col];
          v = 0.5f * v * (1.0f + erff(v * 0.70710678118654752f));
          outh[(size_t)row * N + col] = (_Float16)v;
        } else if (EPI == 2) {
          v += bias[col] + extra[(size_t)row * N + col];
          outf[(size_t)row * N + col] = v;
        } else if (EPI == 3) {
          v += bias[col] + extra[(size_t)(row & (NTOK - 1)) * N + col];
          outf[(size_t)row * N + col] = v;
        } else {
          v += bias[col];
          int b = row >> 10, n = row & (NTOK - 1);
          outf[((size_t)b * N + col) * NTOK + n] = v;
        }
      }
    }
  }
}

// ---------------------------------------------------------------------------
// Flash attention, all-WMMA. One wave = 16 queries of one (batch, head).
// 64 keys per step: 8 WMMAs for S=QK^T, 8 for O+=P*V, 2 for the running
// softmax denominator (P x ones -> row sums, no cross-lane reduction needed).
// V is pre-transposed to vT[b,h,d,key] so P*V B-fragments load contiguously.
// ---------------------------------------------------------------------------
__global__ __launch_bounds__(256) void attn_kernel(
    const _Float16* __restrict__ qkv, const _Float16* __restrict__ vT,
    _Float16* __restrict__ attnout) {
  __shared__ __align__(16) _Float16 plds[8 * 1024];  // 16x64 P tile per wave
  int widx = threadIdx.x >> 5;
  int lane = threadIdx.x & 31;
  int wave = blockIdx.x * 8 + widx;
  int bh = wave >> 6;       // 64 q-tiles per (b,h)
  int qt = wave & 63;
  int b  = bh >> 3;
  int hd = bh & 7;
  int t0 = b * NTOK + qt * 16;

  const _Float16* qp  = qkv + hd * HDIM;
  const _Float16* kp  = qkv + DMODEL + hd * HDIM;
  const _Float16* vtb = vT + (size_t)bh * HDIM * NTOK;
  _Float16* lp = plds + widx * 1024;

  v16h qa0 = load_afrag(qp, 3 * DMODEL, t0, 0);
  v16h qa1 = load_afrag(qp, 3 * DMODEL, t0, 32);

  v16h ones;
#pragma unroll
  for (int i = 0; i < 16; ++i) ones[i] = (_Float16)1.0f;

  v8f zero = {};
  v8f o[5];
#pragma unroll
  for (int i = 0; i < 5; ++i) o[i] = zero;
  float rowmax[8];
#pragma unroll
  for (int r = 0; r < 8; ++r) rowmax[r] = -1e30f;

  const float scale = 0.125f;  // 1/sqrt(64)
  for (int kt = 0; kt < NTOK / 64; ++kt) {
    int kk = b * NTOK + kt * 64;  // key token base (global row)
    v8f s[4];
#pragma unroll
    for (int j = 0; j < 4; ++j) {
      s[j] = wmma32(qa0, load_bfrag(kp, 3 * DMODEL, kk + j * 16, 0), zero);
      s[j] = wmma32(qa1, load_bfrag(kp, 3 * DMODEL, kk + j * 16, 32), s[j]);
    }

    int prow = (lane >> 4) << 3;
#pragma unroll
    for (int r = 0; r < 8; ++r) {
      float a0 = s[0][r] * scale, a1 = s[1][r] * scale;
      float a2 = s[2][r] * scale, a3 = s[3][r] * scale;
      float m = fmaxf(fmaxf(a0, a1), fmaxf(a2, a3));
      m = fmaxf(m, __shfl_xor(m, 1, 32));
      m = fmaxf(m, __shfl_xor(m, 2, 32));
      m = fmaxf(m, __shfl_xor(m, 4, 32));
      m = fmaxf(m, __shfl_xor(m, 8, 32));
      float mn = fmaxf(rowmax[r], m);
      float alpha = __expf(rowmax[r] - mn);
      rowmax[r] = mn;
      o[0][r] *= alpha; o[1][r] *= alpha; o[2][r] *= alpha;
      o[3][r] *= alpha; o[4][r] *= alpha;
      _Float16* lrow = lp + (prow + r) * 64 + (lane & 15);
      lrow[0]  = (_Float16)__expf(a0 - mn);
      lrow[16] = (_Float16)__expf(a1 - mn);
      lrow[32] = (_Float16)__expf(a2 - mn);
      lrow[48] = (_Float16)__expf(a3 - mn);
    }
    __syncthreads();
    v16h pa0 = load_afrag(lp, 64, 0, 0);   // P[:, 0:32]  as A fragment
    v16h pa1 = load_afrag(lp, 64, 0, 32);  // P[:, 32:64] as A fragment
    __syncthreads();

#pragma unroll
    for (int ni = 0; ni < 4; ++ni) {
      o[ni] = wmma32(pa0, load_bfrag(vtb, NTOK, ni * 16, kt * 64), o[ni]);
      o[ni] = wmma32(pa1, load_bfrag(vtb, NTOK, ni * 16, kt * 64 + 32), o[ni]);
    }
    o[4] = wmma32(pa0, ones, o[4]);  // running row sums of P
    o[4] = wmma32(pa1, ones, o[4]);
  }

#pragma unroll
  for (int r = 0; r < 8; ++r) {
    float inv = 1.0f / o[4][r];  // every lane holds its row's sum
    int row = t0 + ((lane >> 4) << 3) + r;
    _Float16* op = attnout + (size_t)row * DMODEL + hd * HDIM;
    op[(lane & 15)]      = (_Float16)(o[0][r] * inv);
    op[16 + (lane & 15)] = (_Float16)(o[1][r] * inv);
    op[32 + (lane & 15)] = (_Float16)(o[2][r] * inv);
    op[48 + (lane & 15)] = (_Float16)(o[3][r] * inv);
  }
}

// ---------------------------------------------------------------------------
// V transpose: qkv16[token][1024 + h*64 + d] -> vT[((b*8+h)*64+d)*1024 + key]
// ---------------------------------------------------------------------------
__global__ void vt_kernel(const _Float16* __restrict__ qkv,
                          _Float16* __restrict__ vT) {
  int idx = blockIdx.x * 256 + threadIdx.x;  // 4096 * 1024 total
  int key = idx & (NTOK - 1);
  int rowd = idx >> 10;              // bh*64 + d  in [0,4096)
  int bh = rowd >> 6, d = rowd & 63;
  int b = bh >> 3, h = bh & 7;
  vT[idx] = qkv[(size_t)(b * NTOK + key) * (3 * DMODEL) + 2 * DMODEL + h * HDIM + d];
}

// ---------------------------------------------------------------------------
// LayerNorm: one 256-thread block per row of h[8192,512] -> f16 output
// ---------------------------------------------------------------------------
__global__ __launch_bounds__(256) void ln_kernel(
    const float* __restrict__ h, const float* __restrict__ g,
    const float* __restrict__ bta, _Float16* __restrict__ y) {
  const int D = DMODEL;
  int row = blockIdx.x;
  const float* x = h + (size_t)row * D;
  float s = 0.f, ss = 0.f;
  for (int c = threadIdx.x; c < D; c += 256) {
    float v = x[c];
    s += v; ss += v * v;
  }
#pragma unroll
  for (int m = 16; m >= 1; m >>= 1) {
    s += __shfl_xor(s, m, 32);
    ss += __shfl_xor(ss, m, 32);
  }
  __shared__ float r1[8], r2[8];
  __shared__ float mu_s, rsig_s;
  int w = threadIdx.x >> 5;
  if ((threadIdx.x & 31) == 0) { r1[w] = s; r2[w] = ss; }
  __syncthreads();
  if (threadIdx.x == 0) {
    float a = 0.f, q = 0.f;
    for (int i = 0; i < 8; ++i) { a += r1[i]; q += r2[i]; }
    float mu = a / D;
    mu_s = mu;
    rsig_s = rsqrtf(q / D - mu * mu + 1e-5f);
  }
  __syncthreads();
  float mu = mu_s, rsig = rsig_s;
  for (int c = threadIdx.x; c < D; c += 256)
    y[(size_t)row * D + c] = (_Float16)((x[c] - mu) * rsig * g[c] + bta[c]);
}

// ---------------------------------------------------------------------------
// Small utility kernels
// ---------------------------------------------------------------------------
__global__ void f32_to_f16_kernel(const float* __restrict__ in,
                                  _Float16* __restrict__ out, int n) {
  int i = blockIdx.x * 256 + threadIdx.x;
  if (i < n) out[i] = (_Float16)in[i];
}

// x[B,CIN,H,W] f32 -> xf16[token=b*1024+hw][c] f16
__global__ void xcvt_kernel(const float* __restrict__ x,
                            _Float16* __restrict__ xf16) {
  int idx = blockIdx.x * 256 + threadIdx.x;  // 8192*256 total
  int t = idx >> 8, c = idx & 255;
  int b = t >> 10, n = t & (NTOK - 1);
  xf16[idx] = (_Float16)x[((size_t)b * CIN + c) * NTOK + n];
}

// DETR 2D sinusoidal positional encoding -> pos[1024][512] f32
__global__ void pos_kernel(float* __restrict__ pos) {
  int idx = blockIdx.x * 256 + threadIdx.x;  // 1024*512
  int n = idx >> 9, c = idx & 511;
  int yi = n >> 5, xi = n & 31;
  const float scale = 6.2831853071795864f;
  const float denom = 32.0f + 1e-6f;
  float coord = (c < 256) ? ((yi + 1.0f) / denom * scale)
                          : ((xi + 1.0f) / denom * scale);
  int j = (c < 256) ? c : (c - 256);
  float t = __powf(10000.0f, (float)(j & ~1) / 256.0f);
  float v = coord / t;
  pos[idx] = (j & 1) ? __cosf(v) : __sinf(v);
}

// ---------------------------------------------------------------------------
// Host launcher
// ---------------------------------------------------------------------------
extern "C" void kernel_launch(void* const* d_in, const int* in_sizes, int n_in,
                              void* d_out, int out_size, void* d_ws,
                              size_t ws_size, hipStream_t stream) {
  const float* x      = (const float*)d_in[0];
  const float* W_in   = (const float*)d_in[1];
  const float* b_in   = (const float*)d_in[2];
  const float* ln1_g  = (const float*)d_in[3];
  const float* ln1_b  = (const float*)d_in[4];
  const float* Wqkv   = (const float*)d_in[5];
  const float* Wproj  = (const float*)d_in[6];
  const float* bproj  = (const float*)d_in[7];
  const float* ln2_g  = (const float*)d_in[8];
  const float* ln2_b  = (const float*)d_in[9];
  const float* W1     = (const float*)d_in[10];
  const float* b1     = (const float*)d_in[11];
  const float* W2     = (const float*)d_in[12];
  const float* b2     = (const float*)d_in[13];
  const float* lnf_g  = (const float*)d_in[14];
  const float* lnf_b  = (const float*)d_in[15];
  const float* W_out  = (const float*)d_in[16];
  const float* b_out  = (const float*)d_in[17];
  float* out = (float*)d_out;

  // workspace layout (bytes, 256-aligned)
  char* ws = (char*)d_ws;
  size_t off = 0;
  auto alloc = [&](size_t bytes) {
    char* p = ws + off;
    off += (bytes + 255) & ~(size_t)255;
    return p;
  };
  float*    h       = (float*)alloc((size_t)MTOT * DMODEL * 4);
  float*    pos     = (float*)alloc((size_t)NTOK * DMODEL * 4);
  _Float16* y16     = (_Float16*)alloc((size_t)MTOT * DMODEL * 2);
  _Float16* xf16    = (_Float16*)alloc((size_t)MTOT * CIN * 2);
  _Float16* qkv16   = (_Float16*)alloc((size_t)MTOT * 3 * DMODEL * 2);
  _Float16* vT      = (_Float16*)alloc((size_t)MTOT * DMODEL * 2);
  _Float16* attn16  = (_Float16*)alloc((size_t)MTOT * DMODEL * 2);
  _Float16* mlp16   = (_Float16*)alloc((size_t)MTOT * HIDDEN * 2);
  _Float16* Win16   = (_Float16*)alloc((size_t)DMODEL * CIN * 2);
  _Float16* Wqkv16  = (_Float16*)alloc((size_t)NDEPTH * 3 * DMODEL * DMODEL * 2);
  _Float16* Wproj16 = (_Float16*)alloc((size_t)NDEPTH * DMODEL * DMODEL * 2);
  _Float16* W116    = (_Float16*)alloc((size_t)NDEPTH * HIDDEN * DMODEL * 2);
  _Float16* W216    = (_Float16*)alloc((size_t)NDEPTH * DMODEL * HIDDEN * 2);
  _Float16* Wout16  = (_Float16*)alloc((size_t)NCOUT * DMODEL * 2);

  auto cvt = [&](const float* src, _Float16* dst, int n) {
    f32_to_f16_kernel<<<(n + 255) / 256, 256, 0, stream>>>(src, dst, n);
  };
  cvt(W_in, Win16, DMODEL * CIN);
  cvt(Wqkv, Wqkv16, NDEPTH * 3 * DMODEL * DMODEL);
  cvt(Wproj, Wproj16, NDEPTH * DMODEL * DMODEL);
  cvt(W1, W116, NDEPTH * HIDDEN * DMODEL);
  cvt(W2, W216, NDEPTH * DMODEL * HIDDEN);
  cvt(W_out, Wout16, NCOUT * DMODEL);

  pos_kernel<<<(NTOK * DMODEL) / 256, 256, 0, stream>>>(pos);
  xcvt_kernel<<<(MTOT * CIN) / 256, 256, 0, stream>>>(x, xf16);

  auto gblocks = [](int M, int N) { return (M / 64) * (N / 64) / 8; };

  // input projection: h = xf16 * W_in^T + b_in + pos
  gemm_kernel<3, true><<<gblocks(MTOT, DMODEL), 256, 0, stream>>>(
      xf16, Win16, b_in, pos, nullptr, h, MTOT, DMODEL, CIN);

  for (int i = 0; i < NDEPTH; ++i) {
    ln_kernel<<<MTOT, 256, 0, stream>>>(h, ln1_g + i * DMODEL,
                                        ln1_b + i * DMODEL, y16);
    gemm_kernel<0, true><<<gblocks(MTOT, 3 * DMODEL), 256, 0, stream>>>(
        y16, Wqkv16 + (size_t)i * 3 * DMODEL * DMODEL, nullptr, nullptr,
        qkv16, nullptr, MTOT, 3 * DMODEL, DMODEL);
    vt_kernel<<<(MTOT * DMODEL) / 256, 256, 0, stream>>>(qkv16, vT);
    attn_kernel<<<(NBATCH * NHEAD * (NTOK / 16)) / 8, 256, 0, stream>>>(
        qkv16, vT, attn16);
    gemm_kernel<2, true><<<gblocks(MTOT, DMODEL), 256, 0, stream>>>(
        attn16, Wproj16 + (size_t)i * DMODEL * DMODEL, bproj + i * DMODEL, h,
        nullptr, h, MTOT, DMODEL, DMODEL);
    ln_kernel<<<MTOT, 256, 0, stream>>>(h, ln2_g + i * DMODEL,
                                        ln2_b + i * DMODEL, y16);
    gemm_kernel<1, true><<<gblocks(MTOT, HIDDEN), 256, 0, stream>>>(
        y16, W116 + (size_t)i * HIDDEN * DMODEL, b1 + i * HIDDEN, nullptr,
        mlp16, nullptr, MTOT, HIDDEN, DMODEL);
    gemm_kernel<2, true><<<gblocks(MTOT, DMODEL), 256, 0, stream>>>(
        mlp16, W216 + (size_t)i * DMODEL * HIDDEN, b2 + i * DMODEL, h, nullptr,
        h, MTOT, DMODEL, HIDDEN);
  }

  ln_kernel<<<MTOT, 256, 0, stream>>>(h, lnf_g, lnf_b, y16);
  gemm_kernel<4, false><<<gblocks(MTOT, NCOUT), 256, 0, stream>>>(
      y16, Wout16, b_out, nullptr, nullptr, out, MTOT, NCOUT, DMODEL);
}
